// MS_BACL_12017318494596
// MI455X (gfx1250) — matrix operand order repaired and checked by reading, not compiled, not run on hardware
//
#include <hip/hip_runtime.h>
#include <math.h>

// ---------------------------------------------------------------------------
// CDNA5 (gfx1250) wave32 WMMA types
// ---------------------------------------------------------------------------
typedef __attribute__((ext_vector_type(16))) __bf16 v16bf;
typedef __attribute__((ext_vector_type(8)))  float  v8f;

// ---------------------------------------------------------------------------
// Elementwise / utility kernels
// ---------------------------------------------------------------------------
__global__ void zero_f32(float* __restrict__ p, size_t n) {
  size_t i = (size_t)blockIdx.x * blockDim.x + threadIdx.x;
  size_t st = (size_t)gridDim.x * blockDim.x;
  for (; i < n; i += st) p[i] = 0.0f;
}

// GIN aggregation: xa[dst[e]] += x[src[e]]  (xa pre-initialized with x -> (x+agg))
__global__ void edge_agg(const float* __restrict__ x, const int* __restrict__ src,
                         const int* __restrict__ dst, float* __restrict__ xa,
                         int E, int F) {
  size_t total = (size_t)E * F;
  size_t i = (size_t)blockIdx.x * blockDim.x + threadIdx.x;
  size_t st = (size_t)gridDim.x * blockDim.x;
  for (; i < total; i += st) {
    int e = (int)(i / F);
    int f = (int)(i % F);
    atomicAdd(&xa[(size_t)dst[e] * F + f], x[(size_t)src[e] * F + f]);
  }
}

// Pack fp32 weight [K,N] into bf16 B-operand fragments (ISA 7.12.2 layout):
// fragment(kc,ct): lane holds col n = ct*16+(lane&15), K = kc*32+(lane>>4)*8+(e<8 ? e : e+8)
__global__ void pack_w_bf16(const float* __restrict__ W, __bf16* __restrict__ Wp,
                            int K, int N, int ntiles, int kchunks) {
  size_t total = (size_t)kchunks * ntiles * 512;
  size_t i = (size_t)blockIdx.x * blockDim.x + threadIdx.x;
  size_t st = (size_t)gridDim.x * blockDim.x;
  for (; i < total; i += st) {
    int frag   = (int)(i >> 9);
    int within = (int)(i & 511);
    int lane = within >> 4, e = within & 15;
    int kc = frag / ntiles, ct = frag % ntiles;
    int k = kc * 32 + ((lane >> 4) << 3) + ((e < 8) ? e : e + 8);
    int n = ct * 16 + (lane & 15);
    float v = (k < K && n < N) ? W[(size_t)k * N + n] : 0.0f;
    Wp[i] = (__bf16)v;
  }
}

// Pack fp32 activation [M,K] into bf16 A-operand fragments (ISA 7.12.2 layout):
// fragment(rt,kc): lane holds row M = rt*16+(lane&15),
//                  K = kc*32 + (lane>>4)*8 + (e<8 ? e : e+8)
__global__ void pack_a_bf16(const float* __restrict__ A, __bf16* __restrict__ Ap,
                            int M, int K, int rowtiles, int kchunks) {
  size_t total = (size_t)rowtiles * kchunks * 512;
  size_t i = (size_t)blockIdx.x * blockDim.x + threadIdx.x;
  size_t st = (size_t)gridDim.x * blockDim.x;
  for (; i < total; i += st) {
    int frag   = (int)(i >> 9);
    int within = (int)(i & 511);
    int lane = within >> 4, e = within & 15;
    int rt = frag / kchunks, kc = frag % kchunks;
    int row = rt * 16 + (lane & 15);
    int k = kc * 32 + ((lane >> 4) << 3) + ((e < 8) ? e : e + 8);
    float v = (row < M && k < K) ? A[(size_t)row * K + k] : 0.0f;
    Ap[i] = (__bf16)v;
  }
}

// ---------------------------------------------------------------------------
// WMMA GEMM on pre-packed bf16 fragments: out = act(A @ W + bias)
//   act: 0=none, 1=relu, 2=sigmoid
//   If gsum/gmax != null: fused graph mean/max pooling epilogue (out skipped).
//   Block = 128 = 4 wave32; each wave owns 4 adjacent 16x16 C tiles, so one
//   A-fragment load (32B/lane) feeds 4 WMMAs. No LDS, no barriers.
// ---------------------------------------------------------------------------
__global__ __launch_bounds__(128) void wmma_gemm_bf16(
    const v16bf* __restrict__ Ap, const v16bf* __restrict__ Wp,
    const float* __restrict__ bias,
    float* __restrict__ out,
    float* __restrict__ gsum, int* __restrict__ gmax,
    const int* __restrict__ batch,
    int M, int N, int ntiles, int kchunks, int act) {
  const int wave = (int)threadIdx.x >> 5;
  const int lane = (int)threadIdx.x & 31;
  const int rt   = blockIdx.x;
  const int ct0  = blockIdx.y * 16 + wave * 4;
  // Wave-uniform activity flags -> scalar branches, EXEC stays all-1s for WMMA.
  const bool a0 = (ct0 + 0) < ntiles;
  const bool a1 = (ct0 + 1) < ntiles;
  const bool a2 = (ct0 + 2) < ntiles;
  const bool a3 = (ct0 + 3) < ntiles;
  if (!a0) return;  // whole wave idle

  const v16bf* ap = Ap + (size_t)rt * kchunks * 32 + lane;
  v8f c0 = {}, c1 = {}, c2 = {}, c3 = {};
  for (int kc = 0; kc < kchunks; ++kc) {
    v16bf af = ap[(size_t)kc * 32];
    if (kc + 1 < kchunks)
      __builtin_prefetch((const void*)&ap[(size_t)(kc + 1) * 32], 0, 1);
    const v16bf* wrow = Wp + (size_t)kc * ntiles * 32 + lane;
    {
      v16bf bf = wrow[(size_t)(ct0 + 0) * 32];
      c0 = __builtin_amdgcn_wmma_f32_16x16x32_bf16(false, af, false, bf,
                                                   (short)0, c0, false, false);
    }
    if (a1) {
      v16bf bf = wrow[(size_t)(ct0 + 1) * 32];
      c1 = __builtin_amdgcn_wmma_f32_16x16x32_bf16(false, af, false, bf,
                                                   (short)0, c1, false, false);
    }
    if (a2) {
      v16bf bf = wrow[(size_t)(ct0 + 2) * 32];
      c2 = __builtin_amdgcn_wmma_f32_16x16x32_bf16(false, af, false, bf,
                                                   (short)0, c2, false, false);
    }
    if (a3) {
      v16bf bf = wrow[(size_t)(ct0 + 3) * 32];
      c3 = __builtin_amdgcn_wmma_f32_16x16x32_bf16(false, af, false, bf,
                                                   (short)0, c3, false, false);
    }
  }

  // Epilogue: C/D layout (ISA 7.12.2): VGPR r -> row = rt*16 + r + 8*(lane>>4),
  // col = ct*16 + (lane&15).
  const int colo = lane & 15;
  const int rbase = rt * 16 + ((lane >> 4) << 3);
  v8f accs[4] = {c0, c1, c2, c3};
  bool as[4] = {a0, a1, a2, a3};
#pragma unroll
  for (int j = 0; j < 4; ++j) {
    if (!as[j]) continue;
    int col = (ct0 + j) * 16 + colo;
    if (col >= N) continue;
    float bv = bias[col];
#pragma unroll
    for (int r = 0; r < 8; ++r) {
      int row = rbase + r;
      if (row >= M) continue;
      float v = accs[j][r] + bv;
      if (act == 1)      v = fmaxf(v, 0.0f);
      else if (act == 2) v = 1.0f / (1.0f + __expf(-v));
      if (out) out[(size_t)row * N + col] = v;
      if (gsum) {  // fused segment sum/max pooling (values >= 0 post-relu)
        int b = batch[row];
        atomicAdd(&gsum[(size_t)b * N + col], v);
        atomicMax(&gmax[(size_t)b * N + col], __float_as_int(v));
      }
    }
  }
}

// ---------------------------------------------------------------------------
// Pooling finish, PairNorm, BatchNorm, head helpers
// ---------------------------------------------------------------------------
__global__ void node_count(const int* __restrict__ batch, float* __restrict__ gcnt, int N) {
  int i = blockIdx.x * blockDim.x + threadIdx.x;
  if (i < N) atomicAdd(&gcnt[batch[i]], 1.0f);
}

__global__ void pool_finish(const float* __restrict__ gsum, const int* __restrict__ gmax,
                            const float* __restrict__ gcnt, float* __restrict__ pooled,
                            int Bg, int C) {
  size_t total = (size_t)Bg * C;
  size_t i = (size_t)blockIdx.x * blockDim.x + threadIdx.x;
  size_t st = (size_t)gridDim.x * blockDim.x;
  for (; i < total; i += st) {
    int b = (int)(i / C), c = (int)(i % C);
    float cnt = fmaxf(gcnt[b], 1.0f);
    pooled[(size_t)b * (2 * C) + c]     = gsum[i] / cnt;
    pooled[(size_t)b * (2 * C) + C + c] = __int_as_float(gmax[i]);
  }
}

__global__ void col_mean(const float* __restrict__ X, float* __restrict__ mu, int R, int C) {
  int j = blockIdx.x * blockDim.x + threadIdx.x;
  if (j >= C) return;
  float s = 0.0f;
  for (int r = 0; r < R; ++r) s += X[(size_t)r * C + j];
  mu[j] = s / (float)R;
}

__global__ void pn_ssq(const float* __restrict__ X, const float* __restrict__ mu,
                       float* __restrict__ scalar, int R, int C) {
  int j = blockIdx.x * blockDim.x + threadIdx.x;
  if (j >= C) return;
  float m = mu[j], s = 0.0f;
  for (int r = 0; r < R; ++r) {
    float d = X[(size_t)r * C + j] - m;
    s += d * d;
  }
  atomicAdd(scalar, s);
}

__global__ void pn_apply(float* __restrict__ X, const float* __restrict__ mu,
                         const float* __restrict__ scalar, int R, int C) {
  size_t total = (size_t)R * C;
  float scale = rsqrtf(1e-5f + scalar[0] / (float)R);
  size_t i = (size_t)blockIdx.x * blockDim.x + threadIdx.x;
  size_t st = (size_t)gridDim.x * blockDim.x;
  for (; i < total; i += st) {
    int j = (int)(i % C);
    X[i] = (X[i] - mu[j]) * scale;
  }
}

__global__ void bn_stats(const float* __restrict__ X, float* __restrict__ mu,
                         float* __restrict__ var, int R, int C) {
  int j = blockIdx.x * blockDim.x + threadIdx.x;
  if (j >= C) return;
  float s = 0.0f, s2 = 0.0f;
  for (int r = 0; r < R; ++r) {
    float v = X[(size_t)r * C + j];
    s += v; s2 += v * v;
  }
  float m = s / (float)R;
  mu[j] = m;
  var[j] = s2 / (float)R - m * m;
}

__global__ void bn_apply(const float* __restrict__ X, const float* __restrict__ mu,
                         const float* __restrict__ var, const float* __restrict__ gam,
                         const float* __restrict__ bet, float* __restrict__ Y,
                         int R, int C) {
  size_t total = (size_t)R * C;
  size_t i = (size_t)blockIdx.x * blockDim.x + threadIdx.x;
  size_t st = (size_t)gridDim.x * blockDim.x;
  for (; i < total; i += st) {
    int j = (int)(i % C);
    Y[i] = (X[i] - mu[j]) * rsqrtf(var[j] + 1e-5f) * gam[j] + bet[j];
  }
}

__global__ void extract_ab(const float* __restrict__ z2, float* __restrict__ alpha,
                           float* __restrict__ beta, int Bg) {
  int i = blockIdx.x * blockDim.x + threadIdx.x;
  if (i < Bg) { alpha[i] = z2[2 * i]; beta[i] = z2[2 * i + 1]; }
}

// ---------------------------------------------------------------------------
// Host launch
// ---------------------------------------------------------------------------
static inline unsigned grid_for(size_t n) {
  size_t b = (n + 255) / 256;
  if (b > 1048576) b = 1048576;
  if (b == 0) b = 1;
  return (unsigned)b;
}

extern "C" void kernel_launch(void* const* d_in, const int* in_sizes, int n_in,
                              void* d_out, int out_size, void* d_ws, size_t ws_size,
                              hipStream_t stream) {
  (void)in_sizes; (void)n_in; (void)out_size; (void)ws_size;
  const int N0 = 100000, N1 = 100000, E = 800000, Bg = 1024;
  const int F0 = 93, F1 = 43;

  const float* x0  = (const float*)d_in[0];
  const int*   ei0 = (const int*)d_in[1];
  const int*   b0  = (const int*)d_in[2];
  const float* x1  = (const float*)d_in[3];
  const int*   ei1 = (const int*)d_in[4];
  const int*   b1  = (const int*)d_in[5];
  const float* w_c1 = (const float*)d_in[6];  const float* b_c1 = (const float*)d_in[7];
  const float* w_c2 = (const float*)d_in[8];  const float* b_c2 = (const float*)d_in[9];
  const float* w_c3 = (const float*)d_in[10]; const float* b_c3 = (const float*)d_in[11];
  const float* w_c4 = (const float*)d_in[12]; const float* b_c4 = (const float*)d_in[13];
  const float* fg0_w1 = (const float*)d_in[14]; const float* fg0_b1 = (const float*)d_in[15];
  const float* fg0_w2 = (const float*)d_in[16]; const float* fg0_b2 = (const float*)d_in[17];
  const float* fg0_g  = (const float*)d_in[18]; const float* fg0_bt = (const float*)d_in[19];
  const float* fg1_w1 = (const float*)d_in[20]; const float* fg1_b1 = (const float*)d_in[21];
  const float* fg1_w2 = (const float*)d_in[22]; const float* fg1_b2 = (const float*)d_in[23];
  const float* fg1_g  = (const float*)d_in[24]; const float* fg1_bt = (const float*)d_in[25];
  const float* ff0_w1 = (const float*)d_in[26]; const float* ff0_b1 = (const float*)d_in[27];
  const float* ff0_w2 = (const float*)d_in[28]; const float* ff0_b2 = (const float*)d_in[29];
  const float* ff1_w1 = (const float*)d_in[30]; const float* ff1_b1 = (const float*)d_in[31];
  const float* ff1_w2 = (const float*)d_in[32]; const float* ff1_b2 = (const float*)d_in[33];

  // Workspace carve (256B aligned); sized for the larger branch, reused by both.
  char* ws = (char*)d_ws;
  size_t off = 0;
  auto carve = [&](size_t bytes) -> void* {
    void* p = ws + off;
    off = (off + bytes + 255) & ~(size_t)255;
    return p;
  };
  const size_t NFMAX = (size_t)N0 * F0;       // 9.3M floats
  const int CMAX = F0 * 10;                   // 930
  float* xa      = (float*)carve(NFMAX * 4);            // (x + agg) buffer
  float* h1      = (float*)carve(NFMAX * 4);            // conv1 output
  float* gsum    = (float*)carve((size_t)Bg * CMAX * 4);
  int*   gmax    = (int*)  carve((size_t)Bg * CMAX * 4);
  float* gcnt    = (float*)carve((size_t)Bg * 4);
  float* pooled  = (float*)carve((size_t)Bg * 2 * CMAX * 4);
  float* mu      = (float*)carve(2048 * 4);
  float* var     = (float*)carve(2048 * 4);
  float* scalarb = (float*)carve(256);
  float* g1buf   = (float*)carve((size_t)Bg * 1024 * 4);
  float* gpre    = (float*)carve((size_t)Bg * 512 * 4);
  float* z1      = (float*)carve((size_t)Bg * 256 * 4);
  float* z2      = (float*)carve((size_t)Bg * 2 * 4);
  __bf16* wpack  = (__bf16*)carve((size_t)2000000 * 2);   // >= 59*64*512 bf16
  __bf16* apack  = (__bf16*)carve((size_t)10000000 * 2);  // >= 6250*3*512 bf16

  // One GEMM = pack A + pack W into fragment layout + WMMA kernel.
  auto launch_gemm = [&](const float* Ain, const float* W, const float* bias,
                         float* outp, float* gsumP, int* gmaxP, const int* batchP,
                         int M, int K, int Nout, int act) {
    int ntiles = (Nout + 15) / 16;
    int kchunks = (K + 31) / 32;
    int rowtiles = (M + 15) / 16;
    size_t wtot = (size_t)kchunks * ntiles * 512;
    size_t atot = (size_t)rowtiles * kchunks * 512;
    pack_w_bf16<<<grid_for(wtot), 256, 0, stream>>>(W, wpack, K, Nout, ntiles, kchunks);
    pack_a_bf16<<<grid_for(atot), 256, 0, stream>>>(Ain, apack, M, K, rowtiles, kchunks);
    dim3 g(rowtiles, (ntiles + 15) / 16);
    wmma_gemm_bf16<<<g, 128, 0, stream>>>((const v16bf*)apack, (const v16bf*)wpack,
                                          bias, outp, gsumP, gmaxP, batchP,
                                          M, Nout, ntiles, kchunks, act);
  };

  auto run_branch = [&](const float* x, const int* ei, const int* batch,
                        int Nn, int F,
                        const float* w1, const float* bv1,
                        const float* w2, const float* bv2,
                        const float* fgw1, const float* fgb1,
                        const float* fgw2, const float* fgb2,
                        const float* gam, const float* bet,
                        const float* ffw1, const float* ffb1,
                        const float* ffw2, const float* ffb2,
                        float* xg, float* alpha, float* beta) {
    const int C = F * 10;
    const int C2 = 2 * C;
    const int* src = ei;
    const int* dst = ei + E;

    // ---- GIN conv 1: h1 = relu((x + agg(x)) @ w1 + b1)
    hipMemcpyAsync(xa, x, (size_t)Nn * F * 4, hipMemcpyDeviceToDevice, stream);
    edge_agg<<<grid_for((size_t)E * F), 256, 0, stream>>>(x, src, dst, xa, E, F);
    launch_gemm(xa, w1, bv1, h1, nullptr, nullptr, nullptr, Nn, F, F, /*relu*/1);

    // ---- GIN conv 2 fused with pooling: never materialize [Nn, 10F]
    hipMemcpyAsync(xa, h1, (size_t)Nn * F * 4, hipMemcpyDeviceToDevice, stream);
    edge_agg<<<grid_for((size_t)E * F), 256, 0, stream>>>(h1, src, dst, xa, E, F);
    zero_f32<<<grid_for((size_t)Bg * C), 256, 0, stream>>>(gsum, (size_t)Bg * C);
    zero_f32<<<grid_for((size_t)Bg * C), 256, 0, stream>>>((float*)gmax, (size_t)Bg * C);
    zero_f32<<<grid_for(Bg), 256, 0, stream>>>(gcnt, (size_t)Bg);
    launch_gemm(xa, w2, bv2, nullptr, gsum, gmax, batch, Nn, F, C, /*relu*/1);
    node_count<<<grid_for(Nn), 256, 0, stream>>>(batch, gcnt, Nn);
    pool_finish<<<grid_for((size_t)Bg * C), 256, 0, stream>>>(gsum, gmax, gcnt, pooled, Bg, C);

    // ---- PairNorm on pooled [Bg, 2C]
    col_mean<<<(C2 + 255) / 256, 256, 0, stream>>>(pooled, mu, Bg, C2);
    zero_f32<<<1, 64, 0, stream>>>(scalarb, 1);
    pn_ssq<<<(C2 + 255) / 256, 256, 0, stream>>>(pooled, mu, scalarb, Bg, C2);
    pn_apply<<<grid_for((size_t)Bg * C2), 256, 0, stream>>>(pooled, mu, scalarb, Bg, C2);

    // ---- fg head: relu(pooled@W1+b1) -> BN(g@W2+b2) -> xg (into d_out)
    launch_gemm(pooled, fgw1, fgb1, g1buf, nullptr, nullptr, nullptr, Bg, C2, 1024, 1);
    launch_gemm(g1buf, fgw2, fgb2, gpre, nullptr, nullptr, nullptr, Bg, 1024, 512, 0);
    bn_stats<<<(512 + 255) / 256, 256, 0, stream>>>(gpre, mu, var, Bg, 512);
    bn_apply<<<grid_for((size_t)Bg * 512), 256, 0, stream>>>(gpre, mu, var, gam, bet, xg, Bg, 512);

    // ---- ff head: relu(xg@W1+b1) -> sigmoid(z@W2+b2) -> alpha/beta
    launch_gemm(xg, ffw1, ffb1, z1, nullptr, nullptr, nullptr, Bg, 512, 256, 1);
    launch_gemm(z1, ffw2, ffb2, z2, nullptr, nullptr, nullptr, Bg, 256, 2, 2);
    extract_ab<<<grid_for(Bg), 256, 0, stream>>>(z2, alpha, beta, Bg);
  };

  // d_out layout: alpha[1024] | beta[1024] | xg0[1024*512] | xg1[1024*512]
  //               | alpha1[1024] | beta1[1024]
  float* out = (float*)d_out;
  float* alpha0 = out;
  float* beta0  = out + 1024;
  float* xg0    = out + 2048;
  float* xg1    = out + 2048 + (size_t)1024 * 512;
  float* alpha1 = out + 2048 + (size_t)2 * 1024 * 512;
  float* beta1  = alpha1 + 1024;

  run_branch(x0, ei0, b0, N0, F0, w_c1, b_c1, w_c2, b_c2,
             fg0_w1, fg0_b1, fg0_w2, fg0_b2, fg0_g, fg0_bt,
             ff0_w1, ff0_b1, ff0_w2, ff0_b2, xg0, alpha0, beta0);
  run_branch(x1, ei1, b1, N1, F1, w_c3, b_c3, w_c4, b_c4,
             fg1_w1, fg1_b1, fg1_w2, fg1_b2, fg1_g, fg1_bt,
             ff1_w1, ff1_b1, ff1_w2, ff1_b2, xg1, alpha1, beta1);
}
